// HeadAttention_2241972928516
// MI455X (gfx1250) — compile-verified, matrix operand order checked by reading, not run on hardware
//
#include <hip/hip_runtime.h>
#include <hip/hip_bf16.h>

typedef __attribute__((ext_vector_type(16))) __bf16 v16bf;
typedef __attribute__((ext_vector_type(8)))  __bf16 v8bf;
typedef __attribute__((ext_vector_type(4)))  __bf16 v4bf;
typedef __attribute__((ext_vector_type(8)))  float  v8f;

#define BATCH 4
#define SEQ   2048
#define EMB   1024
#define HD    64
#define BS    (BATCH * SEQ)   // 8192 flattened rows

// ---------------------------------------------------------------------------
// CDNA5 async global->LDS copy (ISA 15.18.3 op 98, tracked by ASYNCcnt).
// Copies 16 bytes per lane.  VDST = LDS byte address (low 32 bits of the
// generic pointer = LDS offset per the aperture rules), VADDR = 64-bit global.
// ---------------------------------------------------------------------------
__device__ __forceinline__ void async_ld16(void* ldsDst, const void* gSrc) {
    const uint32_t off = (uint32_t)(uintptr_t)ldsDst;
    asm volatile("global_load_async_to_lds_b128 %0, %1, off"
                 :: "v"(off), "v"(gSrc) : "memory");
}
__device__ __forceinline__ void wait_async() {
    asm volatile("s_wait_asynccnt 0x0" ::: "memory");
}

// ---------------------------------------------------------------------------
// WMMA 16x16x32 bf16 fragment layout (wave32, ISA 7.12.2):
//   lanes 0-15  hold K in {0..7, 16..23}, lanes 16-31 hold K in {8..15, 24..31}
// Tiles are laid out so each fragment's K-run is contiguous: A tiles [row][k],
// B tiles [col][k] (pre-transposed).  Fragment build = two 16-byte loads
// (ds_load_b128 from LDS, global_load_b128 from global bf16 buffers).
// ---------------------------------------------------------------------------
__device__ __forceinline__ v16bf frag16(const __bf16* t, int lane, int ld,
                                        int majorbase, int minorbase) {
    const int mj = majorbase + (lane & 15);
    const int kb = (lane & 16) ? 8 : 0;
    const __bf16* p = t + mj * ld + minorbase + kb;
    v8bf lo = *(const v8bf*)(p);
    v8bf hi = *(const v8bf*)(p + 16);
    v16bf f;
#pragma unroll
    for (int e = 0; e < 8; ++e) { f[e] = lo[e]; f[e + 8] = hi[e]; }
    return f;
}

__device__ __forceinline__ v4bf pack4(float4 v) {
    v4bf b;
    b[0] = (__bf16)v.x; b[1] = (__bf16)v.y; b[2] = (__bf16)v.z; b[3] = (__bf16)v.w;
    return b;
}

// ---------------------------------------------------------------------------
// Kernel 0: one-time weight prep: W[EMB][HD] fp32 -> wT[3][HD][EMB] bf16.
// 128 KB out per matrix; runs once, L2-resident either way.
// ---------------------------------------------------------------------------
__global__ __launch_bounds__(256) void wt_prep_kernel(
    const float* __restrict__ Wq, const float* __restrict__ Wk,
    const float* __restrict__ Wv, __bf16* __restrict__ wT) {
    const int idx = blockIdx.x * 256 + threadIdx.x;   // 3*64*256 threads
    const int k4  = idx & 255;                        // 4-element group along EMB
    const int col = (idx >> 8) & 63;
    const int mat = idx >> 14;
    const float* W = (mat == 0) ? Wq : (mat == 1) ? Wk : Wv;
    v4bf b;
#pragma unroll
    for (int i = 0; i < 4; ++i) b[i] = (__bf16)W[(size_t)(k4 * 4 + i) * HD + col];
    *(v4bf*)(wT + ((size_t)(mat * 64 + col)) * EMB + k4 * 4) = b;
}

// ---------------------------------------------------------------------------
// Kernel 1: fused QKV projection.  grid = BS/16, block = 384 (12 waves).
// Wave w -> (mat = w>>2, ntile = w&3).  Weight tiles arrive via async
// global->LDS b128 from the pre-transposed bf16 wT.  Outputs:
//   q, k: bf16 row-major [BS][HD];  v: bf16 transposed [BATCH][HD][SEQ].
// ---------------------------------------------------------------------------
__global__ __launch_bounds__(384) void qkv_proj_kernel(
    const float* __restrict__ x, const __bf16* __restrict__ wT,
    const float* __restrict__ bq, const float* __restrict__ bk,
    const float* __restrict__ bv,
    __bf16* __restrict__ qbuf, __bf16* __restrict__ kbuf,
    __bf16* __restrict__ vT) {

    __shared__ __bf16 Xt[16 * 32];          // 1 KB, [row][k]
    __shared__ __bf16 WtT[3][64 * 32];      // 12 KB, [col][k] per matrix

    const int rb    = blockIdx.x;
    const int tid   = threadIdx.x;
    const int wave  = tid >> 5;
    const int lane  = tid & 31;
    const int mat   = wave >> 2;            // 0=q 1=k 2=v
    const int ntile = wave & 3;

    v8f acc = {};

    for (int kk = 0; kk < EMB; kk += 32) {
        __syncthreads();
        // async-stage W^T tiles: 32 bytes per thread (2 x b128)
        {
            const int m2     = tid >> 7;        // 0..2
            const int within = tid & 127;
            const int col    = within >> 1;
            const int part   = within & 1;      // 16-element half of the k-run
            const __bf16* src = wT + ((size_t)(m2 * 64 + col)) * EMB + kk + part * 16;
            __bf16* dst = &WtT[m2][col * 32 + part * 16];
            async_ld16(dst, src);
            async_ld16(dst + 8, src + 8);
        }
        // stage X tile (threads 0..127): float4 load -> packed bf16 store
        if (tid < 128) {
            const int r  = tid >> 3;
            const int c4 = (tid & 7) * 4;
            const float* src = x + (size_t)(rb * 16 + r) * EMB + kk + c4;
            *(v4bf*)(Xt + r * 32 + c4) = pack4(*(const float4*)src);
            if (kk + 32 < EMB) __builtin_prefetch(src + 32, 0, 3);
        }
        wait_async();
        __syncthreads();

        v16bf a = frag16(Xt, lane, 32, 0, 0);
        v16bf b = frag16(&WtT[mat][0], lane, 32, ntile * 16, 0);
        acc = __builtin_amdgcn_wmma_f32_16x16x32_bf16(
            false, a, false, b, (short)0, acc, false, false);
    }

    const float* bias = (mat == 0) ? bq : (mat == 1) ? bk : bv;
    const int col   = ntile * 16 + (lane & 15);
    const int rhalf = (lane & 16) ? 8 : 0;
    const float bs  = bias[col];

    if (mat == 2) {
        // v: rows of the C tile are 8 consecutive keys -> one packed b128 store
        const int batch = rb >> 7;                  // SEQ/16 = 128 blocks/batch
        const int key0  = (rb & 127) * 16 + rhalf;
        v8bf pk;
#pragma unroll
        for (int r = 0; r < 8; ++r) pk[r] = (__bf16)(acc[r] + bs);
        *(v8bf*)(vT + ((size_t)(batch * 64 + col)) * SEQ + key0) = pk;
    } else {
        __bf16* dst = (mat == 0) ? qbuf : kbuf;
#pragma unroll
        for (int r = 0; r < 8; ++r) {
            const int row = rb * 16 + rhalf + r;
            dst[(size_t)row * HD + col] = (__bf16)(acc[r] + bs);
        }
    }
}

// ---------------------------------------------------------------------------
// Kernel 2: causal flash attention.  grid = BS/64, block = 128 (4 waves);
// wave w owns 16 query rows; streams 32-key chunks with online softmax.
// K and V^T are already bf16 in the right layouts -> staging is pure
// async global->LDS b128 copies; q fragments are global_load_b128.
// ---------------------------------------------------------------------------
__global__ __launch_bounds__(128) void flash_attn_kernel(
    const __bf16* __restrict__ qbuf, const __bf16* __restrict__ kbuf,
    const __bf16* __restrict__ vT, float* __restrict__ out) {

    __shared__ __bf16 Kt[32 * 64];          // 4 KB, [key][h]
    __shared__ __bf16 VtT[64 * 32];         // 4 KB, [h][key]
    __shared__ __bf16 Pt[4][16 * 32];       // 4 KB, per-wave [qrow][key]

    const int blk      = blockIdx.x;
    const int batch    = blk >> 5;                  // SEQ/64 = 32 blocks/batch
    const int rb_local = blk & 31;
    const size_t batch_row0 = (size_t)batch * SEQ;

    const int tid  = threadIdx.x;
    const int wave = tid >> 5;
    const int lane = tid & 31;
    const int wrow0_local = rb_local * 64 + wave * 16;
    const int half = (lane & 16) ? 8 : 0;

    // q fragments: two global_load_b128 each, no conversion
    const __bf16* qbase = qbuf + (batch_row0 + wrow0_local) * HD;
    v16bf qf[2];
#pragma unroll
    for (int c = 0; c < 2; ++c) qf[c] = frag16(qbase, lane, HD, 0, c * 32);

    float mrow[8], lsum[8];
#pragma unroll
    for (int r = 0; r < 8; ++r) { mrow[r] = -1e30f; lsum[r] = 0.0f; }
    v8f acc[4] = {{}, {}, {}, {}};

    const int nsteps = rb_local * 2 + 2;            // causal bound (block-uniform)

    for (int j = 0; j < nsteps; ++j) {
        const int kloc0 = j * 32;
        __syncthreads();
        // stage K chunk: 4 KB contiguous, 32 B per thread via async b128
        {
            const __bf16* src = kbuf + (batch_row0 + kloc0) * HD + tid * 16;
            __bf16* dst = Kt + tid * 16;
            async_ld16(dst, src);
            async_ld16(dst + 8, src + 8);
            if (j + 1 < nsteps) __builtin_prefetch(src + 32 * HD, 0, 3);
        }
        // stage V^T chunk: 64 rows x 64 B; thread -> (h = tid>>1, 32-B half)
        {
            const int h    = tid >> 1;
            const int part = tid & 1;
            const __bf16* src = vT + ((size_t)(batch * 64 + h)) * SEQ + kloc0 + part * 16;
            __bf16* dst = VtT + h * 32 + part * 16;
            async_ld16(dst, src);
            async_ld16(dst + 8, src + 8);
        }
        wait_async();
        __syncthreads();

        // scores: S[16x32] = q . K^T  (two 16x16 tiles, K=32 x 2 over h)
        v8f st[2];
#pragma unroll
        for (int t = 0; t < 2; ++t) {
            v8f s = {};
#pragma unroll
            for (int c = 0; c < 2; ++c) {
                v16bf kb = frag16(Kt, lane, 64, t * 16, c * 32);
                s = __builtin_amdgcn_wmma_f32_16x16x32_bf16(
                    false, qf[c], false, kb, (short)0, s, false, false);
            }
            st[t] = s;
        }

        // scale 1/sqrt(64), causal mask (finite sentinel avoids inf-inf NaN)
#pragma unroll
        for (int t = 0; t < 2; ++t) {
            const int key = kloc0 + t * 16 + (lane & 15);
#pragma unroll
            for (int r = 0; r < 8; ++r) {
                const int rowq = wrow0_local + half + r;
                const float sv = st[t][r] * 0.125f;
                st[t][r] = (key > rowq) ? -1e30f : sv;
            }
        }

        // online softmax; a score row lives across a 16-lane half
#pragma unroll
        for (int r = 0; r < 8; ++r) {
            float mx = fmaxf(st[0][r], st[1][r]);
#pragma unroll
            for (int m = 1; m < 16; m <<= 1) mx = fmaxf(mx, __shfl_xor(mx, m, 32));
            const float mn   = fmaxf(mrow[r], mx);
            const float corr = __expf(mrow[r] - mn);
            const float p0 = __expf(st[0][r] - mn);
            const float p1 = __expf(st[1][r] - mn);
            st[0][r] = p0; st[1][r] = p1;
            float rs = p0 + p1;
#pragma unroll
            for (int m = 1; m < 16; m <<= 1) rs += __shfl_xor(rs, m, 32);
            lsum[r] = lsum[r] * corr + rs;
            mrow[r] = mn;
#pragma unroll
            for (int t = 0; t < 4; ++t) acc[t][r] *= corr;
        }

        // P tile: C-layout -> bf16 LDS -> A-layout fragment
        __bf16* Pw = &Pt[wave][0];
        const int pc = lane & 15;
#pragma unroll
        for (int t = 0; t < 2; ++t)
#pragma unroll
            for (int r = 0; r < 8; ++r)
                Pw[(half + r) * 32 + t * 16 + pc] = (__bf16)st[t][r];
        __syncthreads();

        v16bf pf = frag16(Pw, lane, 32, 0, 0);
#pragma unroll
        for (int tc = 0; tc < 4; ++tc) {
            v16bf vb = frag16(VtT, lane, 32, tc * 16, 0);
            acc[tc] = __builtin_amdgcn_wmma_f32_16x16x32_bf16(
                false, pf, false, vb, (short)0, acc[tc], false, false);
        }
    }

    // epilogue: out = acc / l  (fp32)
#pragma unroll
    for (int tc = 0; tc < 4; ++tc) {
        const int col = tc * 16 + (lane & 15);
#pragma unroll
        for (int r = 0; r < 8; ++r) {
            const size_t row = batch_row0 + wrow0_local + half + r;
            out[row * HD + col] = acc[tc][r] / lsum[r];
        }
    }
}

extern "C" void kernel_launch(void* const* d_in, const int* in_sizes, int n_in,
                              void* d_out, int out_size, void* d_ws, size_t ws_size,
                              hipStream_t stream) {
    const float* x  = (const float*)d_in[0];
    const float* Wq = (const float*)d_in[1];
    const float* bq = (const float*)d_in[2];
    const float* Wk = (const float*)d_in[3];
    const float* bk = (const float*)d_in[4];
    const float* Wv = (const float*)d_in[5];
    const float* bv = (const float*)d_in[6];
    float* out = (float*)d_out;

    // workspace layout (bf16): q[BS][HD] | k[BS][HD] | vT[B][HD][SEQ] | wT[3][HD][EMB]
    uint8_t* wsb = (uint8_t*)d_ws;
    __bf16* qbuf = (__bf16*)(wsb);
    __bf16* kbuf = (__bf16*)(wsb + (1u << 20));
    __bf16* vT   = (__bf16*)(wsb + (2u << 20));
    __bf16* wT   = (__bf16*)(wsb + (3u << 20));

    wt_prep_kernel<<<dim3(3 * 64 * 256 / 256), 256, 0, stream>>>(Wq, Wk, Wv, wT);

    qkv_proj_kernel<<<dim3(BS / 16), 384, 0, stream>>>(
        x, wT, bq, bk, bv, qbuf, kbuf, vT);

    flash_attn_kernel<<<dim3(BS / 64), 128, 0, stream>>>(qbuf, kbuf, vT, out);
}
